// PositionalEncoding_70695161692713
// MI455X (gfx1250) — matrix operand order repaired and checked by reading, not compiled, not run on hardware
//
#include <hip/hip_runtime.h>
#include <hip/hip_bf16.h>
#include <cstdint>

typedef __attribute__((ext_vector_type(16))) _Float16 v16h;
typedef __attribute__((ext_vector_type(8)))  float    v8f;

#define BATCH 2
#define NPTS  4096
#define NQ    (BATCH * NPTS)     // 8192 queries
#define NS1_  32
#define NS2_  64
#define R1_   0.1f
#define R2_   0.2f

// ---------------------------------------------------------------- helpers

__device__ __forceinline__ v8f zero_v8f() {
    v8f z;
#pragma unroll
    for (int i = 0; i < 8; ++i) z[i] = 0.0f;
    return z;
}

// B-fragment (16x16x32 f16 WMMA): lane holds column n = lane%16,
// halves h -> K = kbase + 8*(lane>=16) + (h<8 ? h : h+8).  w is [C][ldk] f16 (row-major).
__device__ __forceinline__ v16h load_bfrag_lds(const _Float16* w, int ldk,
                                               int cbase, int kbase, int lane) {
    const _Float16* p = w + (cbase + (lane & 15)) * ldk + kbase + ((lane >> 4) << 3);
    v16h b;
#pragma unroll
    for (int h = 0; h < 16; ++h) b[h] = p[(h < 8) ? h : (h + 8)];
    return b;
}

// A-fragment K index pattern: lane<16 -> {0..7,16..23}, lane>=16 -> {8..15,24..31}
// i.e. k = (h<8 ? h : h+8) + 8*(lane>>4)

// ---------------------------------------------------------------- K1: ball query (both radii)

__global__ void __launch_bounds__(256)
ball_query_kernel(const float* __restrict__ pts, int* __restrict__ idx1,
                  int* __restrict__ idx2) {
    int t = blockIdx.x * blockDim.x + threadIdx.x;
    if (t >= NQ) return;
    int b = t >> 12, m = t & (NPTS - 1);
    const float* P = pts + (size_t)b * NPTS * 3;
    __builtin_prefetch(P, 0, 1);                      // global_prefetch_b8 (L2-resident cloud)
    float qx = P[m * 3 + 0], qy = P[m * 3 + 1], qz = P[m * 3 + 2];
    int* o1 = idx1 + (size_t)t * NS1_;
    int* o2 = idx2 + (size_t)t * NS2_;
    int c1 = 0, c2 = 0, f1 = -1, f2 = -1;
    const float r1sq = R1_ * R1_, r2sq = R2_ * R2_;
    for (int n = 0; n < NPTS; ++n) {
        if (c1 == NS1_ && c2 == NS2_) break;
        float dx = P[n * 3 + 0] - qx;
        float dy = P[n * 3 + 1] - qy;
        float dz = P[n * 3 + 2] - qz;
        float d2 = dx * dx + dy * dy + dz * dz;
        if (d2 < r2sq) {
            if (c2 < NS2_) { if (c2 == 0) f2 = n; o2[c2++] = n; }
            if (d2 < r1sq && c1 < NS1_) { if (c1 == 0) f1 = n; o1[c1++] = n; }
        }
    }
    if (f1 < 0) f1 = 0;
    if (f2 < 0) f2 = 0;
    for (int j = c1; j < NS1_; ++j) o1[j] = f1;
    for (int j = c2; j < NS2_; ++j) o2[j] = f2;
}

// ---------------------------------------------------------------- K2: LRF (grouping + rotation)

__device__ void smallest_evec(float a00, float a01, float a02, float a11,
                              float a12, float a22, float& zx, float& zy, float& zz) {
    float p1 = a01 * a01 + a02 * a02 + a12 * a12;
    float q = (a00 + a11 + a22) * (1.0f / 3.0f);
    float lam;
    if (p1 < 1e-20f) {
        lam = fminf(a00, fminf(a11, a22));
    } else {
        float p2 = (a00 - q) * (a00 - q) + (a11 - q) * (a11 - q) +
                   (a22 - q) * (a22 - q) + 2.0f * p1;
        float p = sqrtf(p2 * (1.0f / 6.0f));
        float ip = 1.0f / fmaxf(p, 1e-20f);
        float b00 = (a00 - q) * ip, b01 = a01 * ip, b02 = a02 * ip;
        float b11 = (a11 - q) * ip, b12 = a12 * ip, b22 = (a22 - q) * ip;
        float detB = b00 * (b11 * b22 - b12 * b12) - b01 * (b01 * b22 - b12 * b02) +
                     b02 * (b01 * b12 - b11 * b02);
        float r = fminf(1.0f, fmaxf(-1.0f, 0.5f * detB));
        float phi = acosf(r) * (1.0f / 3.0f);
        lam = q + 2.0f * p * cosf(phi + 2.0943951023931953f);  // + 2*pi/3 -> smallest
    }
    float r0x = a00 - lam, r0y = a01, r0z = a02;
    float r1x = a01, r1y = a11 - lam, r1z = a12;
    float r2x = a02, r2y = a12, r2z = a22 - lam;
    float c0x = r0y * r1z - r0z * r1y, c0y = r0z * r1x - r0x * r1z, c0z = r0x * r1y - r0y * r1x;
    float c1x = r0y * r2z - r0z * r2y, c1y = r0z * r2x - r0x * r2z, c1z = r0x * r2y - r0y * r2x;
    float c2x = r1y * r2z - r1z * r2y, c2y = r1z * r2x - r1x * r2z, c2z = r1x * r2y - r1y * r2x;
    float n0 = c0x * c0x + c0y * c0y + c0z * c0z;
    float n1 = c1x * c1x + c1y * c1y + c1z * c1z;
    float n2 = c2x * c2x + c2y * c2y + c2z * c2z;
    float vx = c0x, vy = c0y, vz = c0z, nb = n0;
    if (n1 > nb) { vx = c1x; vy = c1y; vz = c1z; nb = n1; }
    if (n2 > nb) { vx = c2x; vy = c2y; vz = c2z; nb = n2; }
    if (nb < 1e-30f) { zx = 0.0f; zy = 0.0f; zz = 1.0f; return; }
    float in = rsqrtf(nb);
    zx = vx * in; zy = vy * in; zz = vz * in;
}

__global__ void __launch_bounds__(256)
lrf_kernel(const float* __restrict__ pts, const int* __restrict__ idx,
           _Float16* __restrict__ lrf, int S, float radius) {
    int t = blockIdx.x * blockDim.x + threadIdx.x;
    if (t >= NQ) return;
    int b = t >> 12, m = t & (NPTS - 1);
    const float* P = pts + (size_t)b * NPTS * 3;
    float qx = P[m * 3 + 0], qy = P[m * 3 + 1], qz = P[m * 3 + 2];
    const int* id = idx + (size_t)t * S;
    __builtin_prefetch(id, 0, 1);
    // pass 1: weighted covariance
    float cxx = 0, cxy = 0, cxz = 0, cyy = 0, cyz = 0, czz = 0, wsum = 0;
    for (int s = 0; s < S; ++s) {
        int n = id[s];
        float rx = P[n * 3 + 0] - qx, ry = P[n * 3 + 1] - qy, rz = P[n * 3 + 2] - qz;
        float d = sqrtf(rx * rx + ry * ry + rz * rz);
        float w = fmaxf(radius - d, 0.0f);
        cxx += w * rx * rx; cxy += w * rx * ry; cxz += w * rx * rz;
        cyy += w * ry * ry; cyz += w * ry * rz; czz += w * rz * rz;
        wsum += w;
    }
    float inv = 1.0f / (wsum + 1e-8f);
    cxx *= inv; cxy *= inv; cxz *= inv; cyy *= inv; cyz *= inv; czz *= inv;
    float zx, zy, zz;
    smallest_evec(cxx, cxy, cxz, cyy, cyz, czz, zx, zy, zz);
    // pass 2: sign accumulation + xv (xv is invariant to z sign)
    float sd = 0, xvx = 0, xvy = 0, xvz = 0;
    for (int s = 0; s < S; ++s) {
        int n = id[s];
        float rx = P[n * 3 + 0] - qx, ry = P[n * 3 + 1] - qy, rz = P[n * 3 + 2] - qz;
        float d = sqrtf(rx * rx + ry * ry + rz * rz);
        float w = fmaxf(radius - d, 0.0f);
        float pd = rx * zx + ry * zy + rz * zz;
        sd += pd;
        float px = rx - pd * zx, py = ry - pd * zy, pz = rz - pd * zz;
        float ww = (w * w) * (pd * pd);
        xvx += ww * px; xvy += ww * py; xvz += ww * pz;
    }
    float sgn = (sd > 0.0f) ? -1.0f : 1.0f;
    zx *= sgn; zy *= sgn; zz *= sgn;
    float nv = sqrtf(xvx * xvx + xvy * xvy + xvz * xvz);
    float inx = 1.0f / (nv + 1e-8f);
    xvx *= inx; xvy *= inx; xvz *= inx;
    float yx = zy * xvz - zz * xvy, yy = zz * xvx - zx * xvz, yz = zx * xvy - zy * xvx;
    // pass 3: rotate + write f16 [q][s][3]
    _Float16* out = lrf + (size_t)t * S * 3;
    for (int s = 0; s < S; ++s) {
        int n = id[s];
        float rx = P[n * 3 + 0] - qx, ry = P[n * 3 + 1] - qy, rz = P[n * 3 + 2] - qz;
        float fx = rx * xvx + ry * xvy + rz * xvz;
        float fy = rx * yx + ry * yy + rz * yz;
        float fz = rx * zx + ry * zy + rz * zz;
        out[s * 3 + 0] = (_Float16)fx;
        out[s * 3 + 1] = (_Float16)fy;
        out[s * 3 + 2] = (_Float16)fz;
    }
}

// ---------------------------------------------------------------- K3: shared MLP (WMMA) + maxpool

template <int NS>
__global__ void __launch_bounds__(256)
mlp_branch_kernel(const _Float16* __restrict__ lrf,
                  const float* __restrict__ w1, const float* __restrict__ g1, const float* __restrict__ b1,
                  const float* __restrict__ w2, const float* __restrict__ g2, const float* __restrict__ b2,
                  const float* __restrict__ w3, const float* __restrict__ g3, const float* __restrict__ b3,
                  _Float16* __restrict__ feat, int cbase) {
    __shared__ _Float16 s_w1[32 * 32];    // K padded 3->32
    __shared__ _Float16 s_w2[64 * 32];
    __shared__ _Float16 s_w3[128 * 64];
    __shared__ float    s_sc[224], s_bi[224];   // 32 | 64 | 128
    __shared__ _Float16 act[8][16][64];          // per-wave transpose staging

    int t = threadIdx.x;
    for (int i = t; i < 32 * 32; i += 256) {
        int c = i >> 5, k = i & 31;
        s_w1[i] = (k < 3) ? (_Float16)w1[c * 3 + k] : (_Float16)0.0f;
    }
    for (int i = t; i < 64 * 32; i += 256) s_w2[i] = (_Float16)w2[i];
    for (int i = t; i < 128 * 64; i += 256) s_w3[i] = (_Float16)w3[i];
    const float invbn = rsqrtf(1.0f + 1e-5f);
    if (t < 32)  { s_sc[t]       = g1[t] * invbn; s_bi[t]       = b1[t]; }
    if (t < 64)  { s_sc[32 + t]  = g2[t] * invbn; s_bi[32 + t]  = b2[t]; }
    if (t < 128) { s_sc[96 + t]  = g3[t] * invbn; s_bi[96 + t]  = b3[t]; }
    __syncthreads();

    const int lane = t & 31, wv = t >> 5;
    const int q = blockIdx.x * 8 + wv;
    const int m = lane & 15;
    const int koff = (lane >> 4) * 8;     // A-fragment K offset for upper half-wave
    const int rowoff = (lane >> 4) * 8;   // D-fragment row offset for upper half-wave

    __builtin_prefetch(lrf + (size_t)q * NS * 3, 0, 1);   // this wave's rotated coords

    float mx[8];
#pragma unroll
    for (int i = 0; i < 8; ++i) mx[i] = -1e30f;

    const int RT = NS / 16;
    for (int rt = 0; rt < RT; ++rt) {
        // ---- layer 1: A = rotated coords, K=3 (padded to 32)
        v16h a1;
#pragma unroll
        for (int h = 0; h < 16; ++h) a1[h] = (_Float16)0.0f;
        if (lane < 16) {                               // lanes<16 carry K=0..7,16..23
            const _Float16* p = lrf + ((size_t)q * NS + rt * 16 + lane) * 3;
            a1[0] = p[0]; a1[1] = p[1]; a1[2] = p[2];
        }
#pragma unroll
        for (int nt = 0; nt < 2; ++nt) {
            v16h b = load_bfrag_lds(s_w1, 32, nt * 16, 0, lane);
            v8f d = zero_v8f();
            d = __builtin_amdgcn_wmma_f32_16x16x32_f16(false, a1, false, b, (short)0, d, false, false);
            int c = nt * 16 + m;
            float sc = s_sc[c], bi = s_bi[c];
#pragma unroll
            for (int r = 0; r < 8; ++r)
                act[wv][r + rowoff][c] = (_Float16)fmaxf(d[r] * sc + bi, 0.0f);
        }
        asm volatile("s_wait_dscnt 0" ::: "memory");   // wave-local LDS RAW

        // ---- layer 2: K=32
        v16h a2;
        {
            const _Float16* p = &act[wv][m][0];
#pragma unroll
            for (int h = 0; h < 16; ++h) {
                int kh = (h < 8) ? h : (h + 8);
                a2[h] = p[kh + koff];
            }
        }
#pragma unroll
        for (int nt = 0; nt < 4; ++nt) {
            v16h b = load_bfrag_lds(s_w2, 32, nt * 16, 0, lane);
            v8f d = zero_v8f();
            d = __builtin_amdgcn_wmma_f32_16x16x32_f16(false, a2, false, b, (short)0, d, false, false);
            int c = nt * 16 + m;
            float sc = s_sc[32 + c], bi = s_bi[32 + c];
#pragma unroll
            for (int r = 0; r < 8; ++r)
                act[wv][r + rowoff][c] = (_Float16)fmaxf(d[r] * sc + bi, 0.0f);
        }
        asm volatile("s_wait_dscnt 0" ::: "memory");

        // ---- layer 3: K=64 (2 WMMA k-steps), epilogue fused with maxpool
        v16h a30, a31;
        {
            const _Float16* p = &act[wv][m][0];
#pragma unroll
            for (int h = 0; h < 16; ++h) {
                int kh = (h < 8) ? h : (h + 8);
                a30[h] = p[kh + koff];
                a31[h] = p[32 + kh + koff];
            }
        }
#pragma unroll
        for (int nt = 0; nt < 8; ++nt) {
            v16h b0 = load_bfrag_lds(s_w3, 64, nt * 16, 0, lane);
            v16h b1f = load_bfrag_lds(s_w3, 64, nt * 16, 32, lane);
            v8f d = zero_v8f();
            d = __builtin_amdgcn_wmma_f32_16x16x32_f16(false, a30, false, b0, (short)0, d, false, false);
            d = __builtin_amdgcn_wmma_f32_16x16x32_f16(false, a31, false, b1f, (short)0, d, false, false);
            int c = nt * 16 + m;
            float sc = s_sc[96 + c], bi = s_bi[96 + c];
            float v = -1e30f;
#pragma unroll
            for (int r = 0; r < 8; ++r)
                v = fmaxf(v, fmaxf(d[r] * sc + bi, 0.0f));
            mx[nt] = fmaxf(mx[nt], v);
        }
    }
    // combine the two D row-halves and emit pooled features
#pragma unroll
    for (int nt = 0; nt < 8; ++nt) {
        float o = fmaxf(mx[nt], __shfl_xor(mx[nt], 16, 32));
        if (lane < 16)
            feat[(size_t)q * 256 + cbase + nt * 16 + lane] = (_Float16)o;
    }
}

// ---------------------------------------------------------------- K4: final FC 256 -> 128 (WMMA)
// w3 staged to LDS in f16 once per block (two 32KB k-halves), so B-fragments
// come from ds_load instead of per-wave global f32 loads + converts.

__global__ void __launch_bounds__(256)
fc_kernel(const _Float16* __restrict__ feat, const float* __restrict__ w3,
          const float* __restrict__ b3, float* __restrict__ out) {
    __shared__ _Float16 s_w[128 * 128];   // 32 KB: one k-half of w3, f16
    __shared__ float    s_b[128];

    int t = threadIdx.x;
    int lane = t & 31, wv = t >> 5;
    int qbase = blockIdx.x * 128 + wv * 16;
    int m = lane & 15;
    int koff = (lane >> 4) * 8;
    int rowoff = (lane >> 4) * 8;

    if (t < 128) s_b[t] = b3[t];
    __builtin_prefetch(feat + (size_t)(qbase + m) * 256, 0, 1);

    v8f acc[8];
#pragma unroll
    for (int nt = 0; nt < 8; ++nt) acc[nt] = zero_v8f();

    for (int half = 0; half < 2; ++half) {     // uniform across the block
        __syncthreads();                        // previous half fully consumed
        for (int i = t; i < 128 * 128; i += 256) {
            int c = i >> 7, k = i & 127;
            s_w[i] = (_Float16)w3[(size_t)c * 256 + half * 128 + k];
        }
        __syncthreads();

#pragma unroll
        for (int ks2 = 0; ks2 < 4; ++ks2) {
            int ks = half * 4 + ks2;
            v16h a;
            const _Float16* p = feat + (size_t)(qbase + m) * 256 + ks * 32;
#pragma unroll
            for (int h = 0; h < 16; ++h) {
                int kh = (h < 8) ? h : (h + 8);
                a[h] = p[kh + koff];
            }
#pragma unroll
            for (int nt = 0; nt < 8; ++nt) {
                v16h b = load_bfrag_lds(s_w, 128, nt * 16, ks2 * 32, lane);
                acc[nt] = __builtin_amdgcn_wmma_f32_16x16x32_f16(false, a, false, b, (short)0,
                                                                 acc[nt], false, false);
            }
        }
    }
#pragma unroll
    for (int nt = 0; nt < 8; ++nt) {
        int c = nt * 16 + m;
        float bb = s_b[c];
#pragma unroll
        for (int r = 0; r < 8; ++r) {
            int q = qbase + r + rowoff;
            out[(size_t)q * 128 + c] = acc[nt][r] + bb;
        }
    }
}

// ---------------------------------------------------------------- launcher

extern "C" void kernel_launch(void* const* d_in, const int* in_sizes, int n_in,
                              void* d_out, int out_size, void* d_ws, size_t ws_size,
                              hipStream_t stream) {
    const float* pts = (const float*)d_in[0];
    const float* w1a = (const float*)d_in[1];
    const float* g1a = (const float*)d_in[2];
    const float* bb1a = (const float*)d_in[3];
    const float* w1b = (const float*)d_in[4];
    const float* g1b = (const float*)d_in[5];
    const float* bb1b = (const float*)d_in[6];
    const float* w1c = (const float*)d_in[7];
    const float* g1c = (const float*)d_in[8];
    const float* bb1c = (const float*)d_in[9];
    const float* w2a = (const float*)d_in[10];
    const float* g2a = (const float*)d_in[11];
    const float* bb2a = (const float*)d_in[12];
    const float* w2b = (const float*)d_in[13];
    const float* g2b = (const float*)d_in[14];
    const float* bb2b = (const float*)d_in[15];
    const float* w2c = (const float*)d_in[16];
    const float* g2c = (const float*)d_in[17];
    const float* bb2c = (const float*)d_in[18];
    const float* w3 = (const float*)d_in[19];
    const float* b3 = (const float*)d_in[20];

    // workspace layout (bytes)
    char* ws = (char*)d_ws;
    int*       idx1 = (int*)(ws + 0);                                  // 8192*32*4  = 1 MiB
    int*       idx2 = (int*)(ws + (1u << 20));                         // 8192*64*4  = 2 MiB
    _Float16*  lrf1 = (_Float16*)(ws + (3u << 20));                    // 8192*32*3*2
    _Float16*  lrf2 = (_Float16*)(ws + (3u << 20) + 1572864u);         // 8192*64*3*2
    _Float16*  feat = (_Float16*)(ws + (3u << 20) + 1572864u + 3145728u); // 8192*256*2
    float* out = (float*)d_out;

    ball_query_kernel<<<NQ / 256, 256, 0, stream>>>(pts, idx1, idx2);
    lrf_kernel<<<NQ / 256, 256, 0, stream>>>(pts, idx1, lrf1, NS1_, R1_);
    lrf_kernel<<<NQ / 256, 256, 0, stream>>>(pts, idx2, lrf2, NS2_, R2_);
    mlp_branch_kernel<NS1_><<<NQ / 8, 256, 0, stream>>>(lrf1, w1a, g1a, bb1a, w1b, g1b, bb1b,
                                                        w1c, g1c, bb1c, feat, 0);
    mlp_branch_kernel<NS2_><<<NQ / 8, 256, 0, stream>>>(lrf2, w2a, g2a, bb2a, w2b, g2b, bb2b,
                                                        w2c, g2c, bb2c, feat, 128);
    fc_kernel<<<NQ / 128, 256, 0, stream>>>(feat, w3, b3, out);
}
